// OriginalC3Layer_52407190946233
// MI455X (gfx1250) — matrix operand order repaired and checked by reading, not compiled, not run on hardware
//
#include <hip/hip_runtime.h>

typedef __attribute__((ext_vector_type(16))) _Float16 v16h;
typedef __attribute__((ext_vector_type(4)))  _Float16 v4h;
typedef __attribute__((ext_vector_type(4)))  float    v4f;
typedef __attribute__((ext_vector_type(8)))  float    v8f;

#define C_IN   6
#define H_IN   128
#define W_IN   128
#define C_OUT  16
#define H_OUT  124
#define W_OUT  124
#define KWIN   5
#define KTOT   150   // 6 * 5 * 5
#define KPAD   160   // 5 chunks of 32 (zero-padded tail)
#define NCHUNK 5
#define RAWSZ  (C_IN * KWIN * W_IN)   // 3840 halves

// ---------------------------------------------------------------------------
// Kernel 1: scatter branch weights (w3/w4/w6 + CH3/CH4 channel gathers) into a
// dense [16 x 160] f16 filter, stored directly in a per-lane VGPR layout of
// the 16-bit A-matrix (16x32), contiguous K order (B uses the same), for V_WMMA_F32_16X16X32_F16:
//   lane<16 : M=lane,    half h -> K = h
//   lane>=16: M=lane-16, half h -> K = h + 16
// apack[kc][lane][h], so each lane reads its 16 halves (32B) contiguously.
// ---------------------------------------------------------------------------
__global__ void c3_pack_weights(const float* __restrict__ w3,
                                const float* __restrict__ w4,
                                const float* __restrict__ w6,
                                _Float16* __restrict__ apack) {
    const int CH3t[6][3] = {{0,1,2},{1,2,3},{2,3,4},{3,4,5},{0,4,5},{0,1,5}};
    const int CH4t[9][4] = {{0,1,2,3},{1,2,3,4},{2,3,4,5},{0,3,4,5},{0,1,4,5},
                            {0,1,2,5},{0,1,3,4},{1,2,4,5},{0,2,3,5}};
    for (int idx = threadIdx.x; idx < NCHUNK * 32 * 16; idx += blockDim.x) {
        int kc   = idx >> 9;          // / 512
        int lane = (idx >> 4) & 31;
        int h    = idx & 15;
        int o    = lane & 15;         // output channel (M)
        int klocal = h + ((lane >= 16) ? 16 : 0);
        int K = kc * 32 + klocal;
        float val = 0.0f;
        if (K < KTOT) {
            int c  = K / 25;
            int r  = K % 25;
            int ky = r / 5;
            int kx = r % 5;
            if (o < 6) {
                int j = -1;
                for (int t = 0; t < 3; ++t) if (CH3t[o][t] == c) j = t;
                if (j >= 0) val = w3[((o * 3 + j) * 5 + ky) * 5 + kx];
            } else if (o < 15) {
                int oo = o - 6, j = -1;
                for (int t = 0; t < 4; ++t) if (CH4t[oo][t] == c) j = t;
                if (j >= 0) val = w4[((oo * 4 + j) * 5 + ky) * 5 + kx];
            } else {
                val = w6[(c * 5 + ky) * 5 + kx];
            }
        }
        apack[idx] = (_Float16)val;
    }
}

// ---------------------------------------------------------------------------
// Kernel 2: one block per (batch, output row). 8 waves x 16 pixels = 128 cols.
//   Phase 1: raw 6x5x128 f32 tile -> LDS f16, vectorized (b128 load, b64 store)
//   Phase 2: per-wave im2col [16 px][160 K] in LDS, copying 5-wide kx runs
//   Phase 3: 5 x v_wmma_f32_16x16x32_f16, bias, coalesced non-temporal stores
// B layout here (contiguous K, as in apack): lanes 0-15 hold K=0..15 (N=lane), lanes 16-31 hold
// K=16..31 -> each lane reads 16 consecutive K halves = 32B (two ds_load_b128)
// ---------------------------------------------------------------------------
__global__ void __launch_bounds__(256)
c3_conv_wmma(const float* __restrict__ x,
             const float* __restrict__ b3,
             const float* __restrict__ b4,
             const float* __restrict__ b6,
             const _Float16* __restrict__ apack,
             float* __restrict__ out) {
    __shared__ __align__(32) _Float16 lds[RAWSZ + 8 * 16 * KPAD];
    __shared__ float sbias[16];
    _Float16* rawt = lds;              // [6][5][128] f16
    _Float16* im2  = lds + RAWSZ;      // [8 waves][16 px][160]

    const int y   = blockIdx.x;   // output row
    const int b   = blockIdx.y;   // batch
    const int tid = threadIdx.x;

    if (tid < 16) {
        float bv;
        if (tid < 6)       bv = b3[tid];
        else if (tid < 15) bv = b4[tid - 6];
        else               bv = b6[0];
        sbias[tid] = bv;
    }

    // ---- Phase 1: raw input tile (rows y..y+4, 128 cols, 6 ch), float4 loads
    const float* xb = x + (size_t)b * C_IN * H_IN * W_IN;
    for (int j = tid; j < RAWSZ / 4; j += 256) {        // 960 float4 groups
        int f  = j * 4;
        int c  = f / (KWIN * W_IN);
        int rr = f % (KWIN * W_IN);
        int ky = rr / W_IN;
        int xx = rr % W_IN;
        v4f v = *(const v4f*)(xb + (c * H_IN + (y + ky)) * W_IN + xx);
        *(v4h*)(rawt + f) = __builtin_convertvector(v, v4h);
    }
    __syncthreads();

    // ---- Phase 2: im2col. One task = one (wave, pixel, c, ky): copy the
    // 5-wide kx run with constant offsets. Wave 7's tail pixels (x >= 124) may
    // read a few halves past the raw tile (still inside this shared array);
    // their stores are guarded in phase 3, so no clamp needed here.
    for (int t = tid; t < 8 * 16 * C_IN * KWIN; t += 256) {   // 3840 tasks
        int w   = t / (16 * C_IN * KWIN);     // / 480
        int rem = t % (16 * C_IN * KWIN);
        int m   = rem / (C_IN * KWIN);        // / 30
        int cz  = rem % (C_IN * KWIN);        // c*5 + ky
        const _Float16* src = rawt + cz * W_IN + w * 16 + m;
        _Float16*       dst = im2 + (w * 16 + m) * KPAD + cz * KWIN;
#pragma unroll
        for (int kx = 0; kx < KWIN; ++kx)
            dst[kx] = src[kx];
    }
    // zero-fill the K padding 150..159
    for (int t = tid; t < 8 * 16 * (KPAD - KTOT); t += 256) { // 1280 entries
        int w   = t / (16 * (KPAD - KTOT));
        int rem = t % (16 * (KPAD - KTOT));
        int m   = rem / (KPAD - KTOT);
        int k   = rem % (KPAD - KTOT);
        im2[(w * 16 + m) * KPAD + KTOT + k] = (_Float16)0.0f;
    }
    __syncthreads();

    // ---- Phase 3: WMMA
    const int w    = tid >> 5;      // wave id: pixel tile x0 = 16*w
    const int lane = tid & 31;
    const int n    = lane & 15;     // N = output pixel within tile
    const int hi   = lane >> 4;     // lane group (K half / M half selector)

    v16h aw[NCHUNK];
#pragma unroll
    for (int kc = 0; kc < NCHUNK; ++kc)
        aw[kc] = *(const v16h*)(apack + (kc * 32 + lane) * 16);

    v8f acc = {};
    const _Float16* brow = im2 + (w * 16 + n) * KPAD + hi * 16;
#pragma unroll
    for (int kc = 0; kc < NCHUNK; ++kc) {
        v16h bm = *(const v16h*)(brow + kc * 32);
        acc = __builtin_amdgcn_wmma_f32_16x16x32_f16(
                  /*neg_a=*/false, aw[kc], /*neg_b=*/false, bm,
                  /*c_mod=*/(short)0, acc, /*reuse_a=*/false, /*reuse_b=*/false);
    }

    // ---- Bias + coalesced NT stores (C layout: VGPR r -> M = r + 8*hi, N = lane%16)
    const int xcol = w * 16 + n;
    if (xcol < W_OUT) {
#pragma unroll
        for (int r = 0; r < 8; ++r) {
            int ch = r + hi * 8;
            float v = acc[r] + sbias[ch];
            __builtin_nontemporal_store(
                v, &out[(((size_t)b * C_OUT + ch) * H_OUT + y) * W_OUT + xcol]);
        }
    }
}

extern "C" void kernel_launch(void* const* d_in, const int* in_sizes, int n_in,
                              void* d_out, int out_size, void* d_ws, size_t ws_size,
                              hipStream_t stream) {
    const float* x  = (const float*)d_in[0];
    const float* w3 = (const float*)d_in[1];
    const float* b3 = (const float*)d_in[2];
    const float* w4 = (const float*)d_in[3];
    const float* b4 = (const float*)d_in[4];
    const float* w6 = (const float*)d_in[5];
    const float* b6 = (const float*)d_in[6];
    float* out = (float*)d_out;
    _Float16* apack = (_Float16*)d_ws;   // 5*32*16 f16 = 5120 B

    const int B = in_sizes[0] / (C_IN * H_IN * W_IN);   // 256

    c3_pack_weights<<<1, 256, 0, stream>>>(w3, w4, w6, apack);
    c3_conv_wmma<<<dim3(H_OUT, B), 256, 0, stream>>>(x, b3, b4, b6, apack, out);
}
